// WubuMind_83296595738918
// MI455X (gfx1250) — matrix operand order repaired and checked by reading, not compiled, not run on hardware
//
#include <hip/hip_runtime.h>
#include <hip/hip_bf16.h>
#include <math.h>

// Problem constants (match reference)
#define B_  2
#define N_  2048
#define D_  512
#define H_  8
#define L_  4
#define K_  32
#define V_  256
#define HD_ 64
#define BN_ (B_*N_)
#define MOD_ 4294967296.0f
#define SCALE_ 0.125f  // 1/sqrt(HD)

typedef __attribute__((ext_vector_type(16))) _Float16 v16h;
typedef __attribute__((ext_vector_type(8)))  float    v8f;
typedef _Float16 half_t;

// ---------------------------------------------------------------------------
// f32 -> f16 conversion (grid-stride)
// ---------------------------------------------------------------------------
__global__ __launch_bounds__(256)
void k_cvt_f16(const float* __restrict__ src, half_t* __restrict__ dst, int n) {
  int i = blockIdx.x * blockDim.x + threadIdx.x;
  int stride = gridDim.x * blockDim.x;
  for (; i < n; i += stride) dst[i] = (half_t)src[i];
}

// ---------------------------------------------------------------------------
// Build x0 input: xin[bn, 0:D] = emb[indices[bn]], xin[bn, D:2D] = (h/MOD)*proj
// ---------------------------------------------------------------------------
__global__ __launch_bounds__(256)
void k_build_x0(const int* __restrict__ indices, const int* __restrict__ hashes,
                const float* __restrict__ emb, const float* __restrict__ hash_proj,
                half_t* __restrict__ xin) {
  int bn = blockIdx.x;
  int tok = indices[bn];
  float hv = (float)hashes[bn] / MOD_;
  half_t* row = xin + (size_t)bn * (2 * D_);
  const float* erow = emb + (size_t)tok * D_;
  for (int d = threadIdx.x; d < D_; d += 256) {
    row[d]      = (half_t)erow[d];
    row[D_ + d] = (half_t)(hv * hash_proj[d]);
  }
}

// ---------------------------------------------------------------------------
// WMMA GEMM: C[M,N] = A[M,Kd] (f16, row-major) x B[Kd,N] (f16, row-major)
// EPI bits: 1=+bias  2=+residual  4=gelu(exact)  8=store f16 (else f32)
//
// Wave32, 8 waves/block. Block tile = 128(M) x 64(N); each wave owns a
// 16x64 strip -> 4 accumulators, 4 back-to-back v_wmma per 32-wide K-step
// reusing one A fragment. B chunk (32x64 f16) is staged once per block into
// a DOUBLE-BUFFERED transposed LDS image (padded stride 40 halfs): one
// coalesced 16B global load per thread per chunk, issued for chunk k+1
// before computing chunk k (one barrier per K-step). Each lane's B fragment
// is 16 contiguous halfs -> ds_load_b128 pairs; all four fragments are
// loaded before the WMMA chain so the matrix ops issue back-to-back.
// A/B k order: k = 16*(lane/16) + s (contiguous, same for both); f32 C/D per CDNA5 ISA 7.12.2.
// ---------------------------------------------------------------------------
template<int EPI>
__global__ __launch_bounds__(256)
void k_wmma_gemm(const half_t* __restrict__ A, const half_t* __restrict__ Bm,
                 const float* __restrict__ bias, const float* __restrict__ res,
                 float* __restrict__ Cf, half_t* __restrict__ Ch,
                 int M, int N, int Kd) {
  constexpr int BSTRIDE = 40;                 // padded halfs per transposed row
  __shared__ half_t sBT[2][64 * BSTRIDE];     // B^T chunks: [ncol 0..63][k 0..31]

  const int tid  = threadIdx.x;
  const int lane = tid & 31;
  const int wave = tid >> 5;
  const int cb   = blockIdx.x * 64;           // block col base
  const int rb   = blockIdx.y * 128;          // block row base
  const int mrow = rb + wave * 16 + (lane & 15);
  const int kaOff = (lane >> 4) << 4;         // A: lanes 16-31 start at K=+16
  const int kbOff = (lane >> 4) << 4;         // B: lanes 16-31 start at K=+16

  // cooperative B-loader mapping: 256 threads x 8 halfs = 32x64 chunk
  const int ldr = tid >> 3;                   // k-row within chunk (0..31)
  const int ldc = (tid & 7) * 8;              // col group (0,8,..,56)

  v8f acc[4] = {};

  // prologue: stage chunk 0 into buffer 0
  {
    const half_t* gp = Bm + (size_t)ldr * N + cb + ldc;
    half_t t0[8];
    *(int4*)t0 = *(const int4*)gp;            // one coalesced 16B load
#pragma unroll
    for (int j = 0; j < 8; ++j)
      sBT[0][(ldc + j) * BSTRIDE + ldr] = t0[j];
  }
  int cur = 0;

  for (int k0 = 0; k0 < Kd; k0 += 32) {
    __syncthreads();                          // buf[cur] ready; old reads done

    // issue global load for chunk k0+32 early (overlaps ds + wmma below)
    const bool have_next = (k0 + 32) < Kd;
    half_t tnext[8];
    if (have_next) {
      const half_t* gp = Bm + (size_t)(k0 + 32 + ldr) * N + cb + ldc;
      *(int4*)tnext = *(const int4*)gp;
    }

    // A fragment
    const half_t* ap = A + (size_t)mrow * Kd + k0 + kaOff;
    __builtin_prefetch(ap + 64, 0, 0);        // global_prefetch_b8 two chunks ahead
    v16h a;
#pragma unroll
    for (int t = 0; t < 8; ++t) { a[t] = ap[t]; a[t + 8] = ap[t + 8]; }

    // all four B fragments first, then 4 back-to-back WMMAs
    const half_t* bbase = &sBT[cur][(lane & 15) * BSTRIDE + kbOff];
    v16h b0, b1, b2, b3;
#pragma unroll
    for (int t = 0; t < 16; ++t) {
      b0[t] = bbase[t];
      b1[t] = bbase[16 * BSTRIDE + t];
      b2[t] = bbase[32 * BSTRIDE + t];
      b3[t] = bbase[48 * BSTRIDE + t];
    }
    acc[0] = __builtin_amdgcn_wmma_f32_16x16x32_f16(false, a, false, b0, (short)0, acc[0], false, false);
    acc[1] = __builtin_amdgcn_wmma_f32_16x16x32_f16(false, a, false, b1, (short)0, acc[1], false, false);
    acc[2] = __builtin_amdgcn_wmma_f32_16x16x32_f16(false, a, false, b2, (short)0, acc[2], false, false);
    acc[3] = __builtin_amdgcn_wmma_f32_16x16x32_f16(false, a, false, b3, (short)0, acc[3], false, false);

    // transpose-store next chunk into the other buffer (no barrier needed
    // here: reads of that buffer finished before this iteration's barrier)
    if (have_next) {
#pragma unroll
      for (int j = 0; j < 8; ++j)
        sBT[cur ^ 1][(ldc + j) * BSTRIDE + ldr] = tnext[j];
    }
    cur ^= 1;
  }

  const int rbase = rb + wave * 16 + ((lane >> 4) << 3);
#pragma unroll
  for (int nt = 0; nt < 4; ++nt) {
    const int ncol = cb + nt * 16 + (lane & 15);
    const float bsv = (EPI & 1) ? bias[ncol] : 0.f;
#pragma unroll
    for (int r = 0; r < 8; ++r) {
      const int row = rbase + r;
      float v = acc[nt][r] + bsv;
      if (EPI & 2) v += res[(size_t)row * N + ncol];
      if (EPI & 4) v = 0.5f * v * (1.0f + erff(v * 0.70710678118f));
      if (EPI & 8) Ch[(size_t)row * N + ncol] = (half_t)v;
      else         Cf[(size_t)row * N + ncol] = v;
    }
  }
}

// ---------------------------------------------------------------------------
// LayerNorm row kernel: f32 in -> f16 out
// ---------------------------------------------------------------------------
__global__ __launch_bounds__(256)
void k_layernorm(const float* __restrict__ x, const float* __restrict__ g,
                 const float* __restrict__ be, half_t* __restrict__ out) {
  __shared__ float red[256];
  const int r = blockIdx.x;
  const float* xr = x + (size_t)r * D_;
  float s = 0.f;
  for (int d = threadIdx.x; d < D_; d += 256) s += xr[d];
  red[threadIdx.x] = s; __syncthreads();
  for (int o = 128; o > 0; o >>= 1) {
    if (threadIdx.x < o) red[threadIdx.x] += red[threadIdx.x + o];
    __syncthreads();
  }
  const float mean = red[0] / (float)D_;
  __syncthreads();
  float vs = 0.f;
  for (int d = threadIdx.x; d < D_; d += 256) { float t = xr[d] - mean; vs += t * t; }
  red[threadIdx.x] = vs; __syncthreads();
  for (int o = 128; o > 0; o >>= 1) {
    if (threadIdx.x < o) red[threadIdx.x] += red[threadIdx.x + o];
    __syncthreads();
  }
  const float rstd = rsqrtf(red[0] / (float)D_ + 1e-5f);
  for (int d = threadIdx.x; d < D_; d += 256)
    out[(size_t)r * D_ + d] = (half_t)((xr[d] - mean) * rstd * g[d] + be[d]);
}

// ---------------------------------------------------------------------------
// expmap0: pos = tanh(sc*|v|)*v/(sc*|v|) ; xx = |pos|^2 = tanh^2/c
// ---------------------------------------------------------------------------
__global__ __launch_bounds__(256)
void k_expmap(const float* __restrict__ pt, const float* __restrict__ log_c,
              float* __restrict__ pos, float* __restrict__ xx) {
  __shared__ float red[256];
  const int n = blockIdx.x;
  const float c = expf(*log_c);
  const float sc = sqrtf(c);
  const float* vr = pt + (size_t)n * D_;
  float s = 0.f;
  for (int d = threadIdx.x; d < D_; d += 256) { float t = vr[d]; s += t * t; }
  red[threadIdx.x] = s; __syncthreads();
  for (int o = 128; o > 0; o >>= 1) {
    if (threadIdx.x < o) red[threadIdx.x] += red[threadIdx.x + o];
    __syncthreads();
  }
  const float vn = fmaxf(sqrtf(red[0]), 1e-8f);
  const float th = tanhf(sc * vn);
  const float scale = th / (sc * vn);
  for (int d = threadIdx.x; d < D_; d += 256)
    pos[(size_t)n * D_ + d] = scale * vr[d];
  if (threadIdx.x == 0) xx[n] = th * th / c;
}

// ---------------------------------------------------------------------------
// Poincare KNN: per-row full pairwise distance then iterative top-K min
// (matches jax top_k(-dist): ascending distance, smallest index on ties)
// ---------------------------------------------------------------------------
__global__ __launch_bounds__(256)
void k_knn(const float* __restrict__ pos, const float* __restrict__ xx,
           const float* __restrict__ log_c, float* __restrict__ dists,
           int* __restrict__ idx) {
  __shared__ float sQ[D_];
  __shared__ float sDist[N_];
  __shared__ float redv[256];
  __shared__ int   redi[256];
  const int i = blockIdx.x;
  const float c = expf(*log_c);
  const float rsc = rsqrtf(c);
  for (int d = threadIdx.x; d < D_; d += 256) sQ[d] = pos[(size_t)i * D_ + d];
  __syncthreads();
  const float xi = xx[i];
  for (int j = threadIdx.x; j < N_; j += 256) {
    const float* pj = pos + (size_t)j * D_;
    float dot = 0.f;
    for (int d = 0; d < D_; ++d) dot += sQ[d] * pj[d];
    float d2 = fmaxf(xi + xx[j] - 2.f * dot, 0.f);
    float denom = (1.f - c * xi) * (1.f - c * xx[j]) + 1e-8f;
    float arg = fmaxf(1.f + 2.f * c * d2 / denom, 1.f);
    sDist[j] = acoshf(arg) * rsc;
  }
  __syncthreads();
  for (int kk = 0; kk < K_; ++kk) {
    float bv = INFINITY; int bi = N_;
    for (int j = threadIdx.x; j < N_; j += 256) {
      float v = sDist[j];
      if (v < bv || (v == bv && j < bi)) { bv = v; bi = j; }
    }
    redv[threadIdx.x] = bv; redi[threadIdx.x] = bi; __syncthreads();
    for (int o = 128; o > 0; o >>= 1) {
      if (threadIdx.x < o) {
        float ov = redv[threadIdx.x + o]; int oi = redi[threadIdx.x + o];
        if (ov < redv[threadIdx.x] ||
            (ov == redv[threadIdx.x] && oi < redi[threadIdx.x])) {
          redv[threadIdx.x] = ov; redi[threadIdx.x] = oi;
        }
      }
      __syncthreads();
    }
    if (threadIdx.x == 0) {
      dists[i * K_ + kk] = redv[0];
      idx[i * K_ + kk]   = redi[0];
      sDist[redi[0]] = INFINITY;
    }
    __syncthreads();
  }
}

// ---------------------------------------------------------------------------
// Gathered 32-neighbor attention. Block = one token (b,n); wave h = head h;
// lane = neighbor. Wave-shuffle softmax, then 256 threads produce D outputs.
// ---------------------------------------------------------------------------
__global__ __launch_bounds__(256)
void k_attn(const float* __restrict__ q, const float* __restrict__ kmat,
            const float* __restrict__ vmat, const int* __restrict__ idx,
            const float* __restrict__ dists, const float* __restrict__ log_tau,
            int layer, half_t* __restrict__ ao) {
  __shared__ float sQ[D_];
  __shared__ float sP[H_ * K_];
  __shared__ int   sIdx[K_];
  const int bn = blockIdx.x;
  const int b = bn / N_, n = bn % N_;
  const int lane = threadIdx.x & 31, h = threadIdx.x >> 5;
  for (int d = threadIdx.x; d < D_; d += 256) sQ[d] = q[(size_t)bn * D_ + d];
  if (threadIdx.x < K_) sIdx[threadIdx.x] = idx[n * K_ + threadIdx.x];
  __syncthreads();
  const float tau = expf(log_tau[layer]) + 1e-8f;
  const int j = sIdx[lane];
  const float* kr = kmat + ((size_t)(b * N_ + j)) * D_ + h * HD_;
  float dot = 0.f;
  for (int d = 0; d < HD_; ++d) dot += sQ[h * HD_ + d] * kr[d];
  float logit = dot * SCALE_ - dists[n * K_ + lane] / tau;
  float m = logit;
  for (int o = 16; o > 0; o >>= 1) m = fmaxf(m, __shfl_xor(m, o, 32));
  float e = expf(logit - m);
  float s = e;
  for (int o = 16; o > 0; o >>= 1) s += __shfl_xor(s, o, 32);
  sP[h * K_ + lane] = e / s;
  __syncthreads();
  for (int o = threadIdx.x; o < D_; o += 256) {
    const int hh = o >> 6, d = o & 63;
    float accum = 0.f;
    for (int kk = 0; kk < K_; ++kk)
      accum += sP[hh * K_ + kk] *
               vmat[((size_t)(b * N_ + sIdx[kk])) * D_ + hh * HD_ + d];
    ao[(size_t)bn * D_ + o] = (half_t)accum;
  }
}

// ---------------------------------------------------------------------------
// Output head: out[b, v] = x[b, N-1, :] . Wout[:, v] + bout[v]   (2x256 only)
// ---------------------------------------------------------------------------
__global__ __launch_bounds__(256)
void k_head(const float* __restrict__ x, const float* __restrict__ Wout,
            const float* __restrict__ bout, float* __restrict__ out) {
  const int b = blockIdx.x;
  const int vcol = threadIdx.x;
  const float* xr = x + ((size_t)(b * N_ + N_ - 1)) * D_;
  float acc = bout[vcol];
  for (int d = 0; d < D_; ++d) acc += xr[d] * Wout[d * V_ + vcol];
  out[b * V_ + vcol] = acc;
}

// ---------------------------------------------------------------------------
extern "C" void kernel_launch(void* const* d_in, const int* in_sizes, int n_in,
                              void* d_out, int out_size, void* d_ws, size_t ws_size,
                              hipStream_t stream) {
  const int*   hashes    = (const int*)d_in[0];
  const int*   indices   = (const int*)d_in[1];
  const float* emb       = (const float*)d_in[2];
  const float* hash_proj = (const float*)d_in[3];
  const float* Wb        = (const float*)d_in[4];
  const float* bb        = (const float*)d_in[5];
  const float* log_c     = (const float*)d_in[6];
  const float* pos_t     = (const float*)d_in[7];
  const float* Wq        = (const float*)d_in[8];
  const float* bq        = (const float*)d_in[9];
  const float* Wk        = (const float*)d_in[10];
  const float* bk        = (const float*)d_in[11];
  const float* Wv        = (const float*)d_in[12];
  const float* bv        = (const float*)d_in[13];
  const float* Wo        = (const float*)d_in[14];
  const float* bo        = (const float*)d_in[15];
  const float* W1        = (const float*)d_in[16];
  const float* b1        = (const float*)d_in[17];
  const float* W2        = (const float*)d_in[18];
  const float* b2        = (const float*)d_in[19];
  const float* g1        = (const float*)d_in[20];
  const float* be1       = (const float*)d_in[21];
  const float* g2        = (const float*)d_in[22];
  const float* be2       = (const float*)d_in[23];
  const float* log_tau   = (const float*)d_in[24];
  const float* Wout      = (const float*)d_in[25];
  const float* bout      = (const float*)d_in[26];
  float* out = (float*)d_out;

  // Workspace bump allocator (256B aligned)
  char* ws = (char*)d_ws;
  size_t off = 0;
  auto alloc = [&](size_t bytes) -> void* {
    off = (off + 255) & ~(size_t)255;
    void* p = ws + off;
    off += bytes;
    return p;
  };

  half_t* Wb_h  = (half_t*)alloc((size_t)2 * D_ * D_ * 2);
  half_t* Wq_h  = (half_t*)alloc((size_t)L_ * D_ * D_ * 2);
  half_t* Wk_h  = (half_t*)alloc((size_t)L_ * D_ * D_ * 2);
  half_t* Wv_h  = (half_t*)alloc((size_t)L_ * D_ * D_ * 2);
  half_t* Wo_h  = (half_t*)alloc((size_t)L_ * D_ * D_ * 2);
  half_t* W1_h  = (half_t*)alloc((size_t)L_ * D_ * 4 * D_ * 2);
  half_t* W2_h  = (half_t*)alloc((size_t)L_ * 4 * D_ * D_ * 2);
  half_t* xin_h = (half_t*)alloc((size_t)BN_ * 2 * D_ * 2);
  float*  x     = (float*) alloc((size_t)BN_ * D_ * 4);
  half_t* xn_h  = (half_t*)alloc((size_t)BN_ * D_ * 2);
  float*  qbuf  = (float*) alloc((size_t)BN_ * D_ * 4);
  float*  kbuf  = (float*) alloc((size_t)BN_ * D_ * 4);
  float*  vbuf  = (float*) alloc((size_t)BN_ * D_ * 4);
  half_t* ao_h  = (half_t*)alloc((size_t)BN_ * D_ * 2);
  half_t* h1_h  = (half_t*)alloc((size_t)BN_ * 4 * D_ * 2);
  float*  pos   = (float*) alloc((size_t)N_ * D_ * 4);
  float*  xx    = (float*) alloc((size_t)N_ * 4);
  float*  dists = (float*) alloc((size_t)N_ * K_ * 4);
  int*    idx   = (int*)   alloc((size_t)N_ * K_ * 4);

  auto cvt = [&](const float* src, half_t* dst, size_t n) {
    k_cvt_f16<<<dim3(1024), dim3(256), 0, stream>>>(src, dst, (int)n);
  };

  // --- weight conversion to f16 (one-time per call; fits in 192MB L2) ---
  cvt(Wb, Wb_h, (size_t)2 * D_ * D_);
  cvt(Wq, Wq_h, (size_t)L_ * D_ * D_);
  cvt(Wk, Wk_h, (size_t)L_ * D_ * D_);
  cvt(Wv, Wv_h, (size_t)L_ * D_ * D_);
  cvt(Wo, Wo_h, (size_t)L_ * D_ * D_);
  cvt(W1, W1_h, (size_t)L_ * D_ * 4 * D_);
  cvt(W2, W2_h, (size_t)L_ * 4 * D_ * D_);

  // --- embedding + hash feature build, then x0 = xin @ Wb + bb ---
  k_build_x0<<<dim3(BN_), dim3(256), 0, stream>>>(indices, hashes, emb, hash_proj, xin_h);
  k_wmma_gemm<1><<<dim3(D_ / 64, BN_ / 128), 256, 0, stream>>>(
      xin_h, Wb_h, bb, nullptr, x, nullptr, BN_, D_, 2 * D_);

  // --- Poincare positions + KNN (shared by all layers) ---
  k_expmap<<<dim3(N_), dim3(256), 0, stream>>>(pos_t, log_c, pos, xx);
  k_knn<<<dim3(N_), dim3(256), 0, stream>>>(pos, xx, log_c, dists, idx);

  // --- transformer layers ---
  for (int l = 0; l < L_; ++l) {
    const half_t* Wql = Wq_h + (size_t)l * D_ * D_;
    const half_t* Wkl = Wk_h + (size_t)l * D_ * D_;
    const half_t* Wvl = Wv_h + (size_t)l * D_ * D_;
    const half_t* Wol = Wo_h + (size_t)l * D_ * D_;
    const half_t* W1l = W1_h + (size_t)l * D_ * 4 * D_;
    const half_t* W2l = W2_h + (size_t)l * 4 * D_ * D_;

    k_layernorm<<<dim3(BN_), dim3(256), 0, stream>>>(x, g1 + l * D_, be1 + l * D_, xn_h);

    dim3 gsq(D_ / 64, BN_ / 128);
    k_wmma_gemm<1><<<gsq, 256, 0, stream>>>(xn_h, Wql, bq + l * D_, nullptr, qbuf, nullptr, BN_, D_, D_);
    k_wmma_gemm<1><<<gsq, 256, 0, stream>>>(xn_h, Wkl, bk + l * D_, nullptr, kbuf, nullptr, BN_, D_, D_);
    k_wmma_gemm<1><<<gsq, 256, 0, stream>>>(xn_h, Wvl, bv + l * D_, nullptr, vbuf, nullptr, BN_, D_, D_);

    k_attn<<<dim3(BN_), dim3(256), 0, stream>>>(qbuf, kbuf, vbuf, idx, dists, log_tau, l, ao_h);

    // x = x + ao @ Wo + bo   (bias + residual epilogue)
    k_wmma_gemm<3><<<gsq, 256, 0, stream>>>(ao_h, Wol, bo + l * D_, x, x, nullptr, BN_, D_, D_);

    k_layernorm<<<dim3(BN_), dim3(256), 0, stream>>>(x, g2 + l * D_, be2 + l * D_, xn_h);

    // h1 = gelu(xn2 @ W1 + b1), stored f16 for next GEMM
    k_wmma_gemm<1 | 4 | 8><<<dim3(4 * D_ / 64, BN_ / 128), 256, 0, stream>>>(
        xn_h, W1l, b1 + (size_t)l * 4 * D_, nullptr, nullptr, h1_h, BN_, 4 * D_, D_);

    // x = x + h1 @ W2 + b2
    k_wmma_gemm<3><<<gsq, 256, 0, stream>>>(h1_h, W2l, b2 + l * D_, x, x, nullptr, BN_, D_, 4 * D_);
  }

  // --- output head (only last token per batch) ---
  k_head<<<dim3(B_), dim3(V_), 0, stream>>>(x, Wout, bout, out);
}
